// EquivariantPPFAttention_73014444032168
// MI455X (gfx1250) — compile-verified, matrix operand order checked by reading, not run on hardware
//
#include <hip/hip_runtime.h>
#include <math.h>

// Problem constants (from the reference)
#define NPTS 20000
#define KNBR 32
#define OUT  192
#define EPSV 1e-8f
#define INV_PI 0.3183098861837907f

typedef __attribute__((ext_vector_type(16))) _Float16 v16h;
typedef __attribute__((ext_vector_type(8)))  float    v8f;

static __device__ __forceinline__ v8f wmma_f16(v16h a, v16h b, v8f c) {
    // D = A(16x32 f16) x B(32x16 f16) + C(16x16 f32)
    return __builtin_amdgcn_wmma_f32_16x16x32_f16(
        /*neg_a=*/false, a, /*neg_b=*/false, b,
        /*c_mod=*/(short)0, c, /*reuse_a=*/false, /*reuse_b=*/false);
}

// ---------------------------------------------------------------------------
// Fragment-native LDS storage.
// CDNA5 16-bit A/B fragment (16 rows x 32 K): lane = (m&15) | (k-bit3 << 4),
// in-lane half position = (k-bit4 << 3) | (k&7).  Matrices are stored so each
// lane's 16 halves of a fragment are CONTIGUOUS (32 bytes) -> fragment loads
// are 2x ds_load_b128 instead of 16x ds_load_u16.
// ---------------------------------------------------------------------------
#define FS 16  // halves per lane-slot (16 halves = 32 bytes)

static __device__ __forceinline__ int frag_addr(int m, int k, int NS) {
    int rt = m >> 4, s = k >> 5, kk = k & 31;
    int laneid = (m & 15) | (((kk >> 3) & 1) << 4);
    int pos = ((kk >> 4) << 3) | (kk & 7);
    return ((rt * NS + s) * 32 + laneid) * FS + pos;
}

static __device__ __forceinline__ v16h load_frag(const _Float16* base, int rt, int s,
                                                 int NS, int lane) {
    return *(const v16h*)(base + ((rt * NS + s) * 32 + lane) * FS);
}

struct F3 { float x, y, z; };

static __device__ __forceinline__ float angle_fn(F3 a, F3 b) {
    float y  = a.x * b.x + a.y * b.y + a.z * b.z;
    float cx = a.y * b.z - a.z * b.y;
    float cy = a.z * b.x - a.x * b.z;
    float cz = a.x * b.y - a.y * b.x;
    float xn = sqrtf(cx * cx + cy * cy + cz * cz);
    float sg = (y > 0.f) ? 1.f : ((y < 0.f) ? -1.f : 0.f);
    float xs = (fabsf(y) < EPSV) ? sg * EPSV : y;
    return atan2f(xn, xs) * INV_PI;
}

// ---------------------------------------------------------------------------
// Kernel 1: PPF -> MLP (layer1 VALU, layers 2/3 WMMA) -> mean over K
//           -> gate GEMM via WMMA -> sigmoid.
// 256 threads = 8 waves, one query point per wave; the block's 8 ppf_mod rows
// share one 16x64 gate A-tile (rows 8..15 zero).
// ---------------------------------------------------------------------------
__global__ void __launch_bounds__(256)
ppf_mlp_gate_kernel(const float* __restrict__ q_pts, const float* __restrict__ s_pts,
                    const int* __restrict__ nbr, const float* __restrict__ normals,
                    const float* __restrict__ W1, const float* __restrict__ b1,
                    const float* __restrict__ W2, const float* __restrict__ b2,
                    const float* __restrict__ W3, const float* __restrict__ b3,
                    const float* __restrict__ Wg, const float* __restrict__ bg,
                    float* __restrict__ gate_ws) {
    __shared__ __align__(32) _Float16 sW2[8  * 32 * FS];   // W2^T: 4 ct x 2 s
    __shared__ __align__(32) _Float16 sW3[8  * 32 * FS];   // W3^T: 4 ct x 2 s
    __shared__ __align__(32) _Float16 sWg[24 * 32 * FS];   // Wg^T: 12 ct x 2 s
    __shared__ __align__(32) _Float16 sH[8][4 * 32 * FS];  // per-wave act: 2 rt x 2 s
    __shared__ __align__(32) _Float16 sPA[2 * 32 * FS];    // gate A: 16x64, 2 s
    __shared__ float sW1f[256];                            // W1 (64x4) f32
    __shared__ float sB1f[64], sB2[64], sB3[64];

    const int tid  = threadIdx.x;
    const int lane = tid & 31;
    const int w    = tid >> 5;

    // ---- cooperative staging (f32 -> f16, swizzled into fragment order) ----
    for (int i = tid; i < 64 * 64; i += 256) {
        int nn = i >> 6, k = i & 63;                 // B^T element (k, nn) = W[nn][k]
        sW2[frag_addr(nn, k, 2)] = (_Float16)W2[i];
        sW3[frag_addr(nn, k, 2)] = (_Float16)W3[i];
    }
    for (int i = tid; i < 192 * 64; i += 256) {
        int o = i >> 6, dd = i & 63;
        sWg[frag_addr(o, dd, 2)] = (_Float16)Wg[i];
    }
    for (int i = tid; i < 2 * 32 * FS; i += 256) sPA[i] = (_Float16)0.f;  // pad rows 8..15
    sW1f[tid] = W1[tid];                               // 256 threads == 256 elements
    if (tid < 64) { sB1f[tid] = b1[tid]; sB2[tid] = b2[tid]; sB3[tid] = b3[tid]; }
    __syncthreads();

    // NPTS % 8 == 0, so every wave owns a valid point (barriers below stay safe)
    const int n = blockIdx.x * 8 + w;
    _Float16* sHw = sH[w];
    const int colo = lane & 15;
    const int rofs = (lane < 16) ? 0 : 8;

    // -------- PPF feature row: one neighbor per lane --------
    const int* np  = nbr + (size_t)n * KNBR;
    const int  idx = np[lane];
    const int  qix = np[0];
    F3 q  = { q_pts[n * 3 + 0], q_pts[n * 3 + 1], q_pts[n * 3 + 2] };
    F3 qn = { normals[qix * 3 + 0], normals[qix * 3 + 1], normals[qix * 3 + 2] };
    F3 p  = { s_pts[idx * 3 + 0], s_pts[idx * 3 + 1], s_pts[idx * 3 + 2] };
    F3 nm = { normals[idx * 3 + 0], normals[idx * 3 + 1], normals[idx * 3 + 2] };
    F3 vd = { p.x - q.x, p.y - q.y, p.z - q.z };
    float ppf0 = sqrtf(vd.x * vd.x + vd.y * vd.y + vd.z * vd.z);
    float ppf1 = angle_fn(qn, vd);
    float ppf2 = angle_fn(nm, vd);
    float ppf3 = angle_fn(qn, nm);

    // -------- Layer 1 in VALU (K=4: matrix pipe would run at 1/8 util) ------
    // Weights come from LDS (not hoisted into VGPRs/SGPRs); unroll-by-8 keeps
    // the 8 contiguous fragment-slot stores mergeable into ds_store_b128.
#pragma unroll 8
    for (int c = 0; c < 64; ++c) {
        float v = sB1f[c] + sW1f[c * 4 + 0] * ppf0 + sW1f[c * 4 + 1] * ppf1
                          + sW1f[c * 4 + 2] * ppf2 + sW1f[c * 4 + 3] * ppf3;
        sHw[frag_addr(lane, c, 2)] = (_Float16)fmaxf(v, 0.f);
    }

    // -------- Layer 2: (32x64)@(64x64), 16 WMMA --------
    {
        v16h a[2][2];
#pragma unroll
        for (int r = 0; r < 2; ++r)
#pragma unroll
            for (int s = 0; s < 2; ++s) a[r][s] = load_frag(sHw, r, s, 2, lane);
        v8f acc[2][4];
#pragma unroll
        for (int r = 0; r < 2; ++r)
#pragma unroll
            for (int c = 0; c < 4; ++c) { v8f z = {}; acc[r][c] = z; }
#pragma unroll
        for (int c = 0; c < 4; ++c)
#pragma unroll
            for (int s = 0; s < 2; ++s) {
                v16h bf = load_frag(sW2, c, s, 2, lane);
#pragma unroll
                for (int r = 0; r < 2; ++r) acc[r][c] = wmma_f16(a[r][s], bf, acc[r][c]);
            }
#pragma unroll
        for (int r = 0; r < 2; ++r)
#pragma unroll
            for (int c = 0; c < 4; ++c) {
                float bb = sB2[c * 16 + colo];
#pragma unroll
                for (int i = 0; i < 8; ++i) {
                    float v = fmaxf(acc[r][c][i] + bb, 0.f);
                    sHw[frag_addr(r * 16 + rofs + i, c * 16 + colo, 2)] = (_Float16)v;
                }
            }
    }

    // -------- Layer 3 + mean over the 32 neighbors, 16 WMMA --------
    {
        v16h a[2][2];
#pragma unroll
        for (int r = 0; r < 2; ++r)
#pragma unroll
            for (int s = 0; s < 2; ++s) a[r][s] = load_frag(sHw, r, s, 2, lane);
        v8f acc[2][4];
#pragma unroll
        for (int r = 0; r < 2; ++r)
#pragma unroll
            for (int c = 0; c < 4; ++c) { v8f z = {}; acc[r][c] = z; }
#pragma unroll
        for (int c = 0; c < 4; ++c)
#pragma unroll
            for (int s = 0; s < 2; ++s) {
                v16h bf = load_frag(sW3, c, s, 2, lane);
#pragma unroll
                for (int r = 0; r < 2; ++r) acc[r][c] = wmma_f16(a[r][s], bf, acc[r][c]);
            }
        // column sums via C-fragment structure + one shfl_xor(16)
#pragma unroll
        for (int c = 0; c < 4; ++c) {
            float bb = sB3[c * 16 + colo];
            float ssum = 0.f;
#pragma unroll
            for (int r = 0; r < 2; ++r)
#pragma unroll
                for (int i = 0; i < 8; ++i) ssum += acc[r][c][i] + bb;
            ssum += __shfl_xor(ssum, 16);
            if (lane < 16)  // point w's ppf_mod row into the shared gate A-tile
                sPA[frag_addr(w, c * 16 + lane, 2)] = (_Float16)(ssum * (1.f / 32.f));
        }
    }
    __syncthreads();

    // -------- Gate: sigmoid( ppf_mod(16x64) @ Wg^T(64x192) + bg ), 24 WMMA/block
    {
        v16h aG[2];
#pragma unroll
        for (int s = 0; s < 2; ++s) aG[s] = load_frag(sPA, 0, s, 2, lane);
        for (int ct = w; ct < 12; ct += 8) {
            v8f acc = {};
#pragma unroll
            for (int s = 0; s < 2; ++s)
                acc = wmma_f16(aG[s], load_frag(sWg, ct, s, 2, lane), acc);
            if (lane < 16) {              // rows 0..7 = this block's 8 points
                int col = ct * 16 + lane;
                float bgv = bg[col];
                int nbase = blockIdx.x * 8;
#pragma unroll
                for (int i = 0; i < 8; ++i) {
                    float g = 1.f / (1.f + expf(-(acc[i] + bgv)));
                    gate_ws[(size_t)(nbase + i) * OUT + col] = g;
                }
            }
        }
    }
}

// ---------------------------------------------------------------------------
// Kernel 2: gather+mean s_feats (the 164 MB HBM stream), Wv transform via
// WMMA, gate scaling.  16 points/block: out(192x48) = Wv(192x64) @ agg(64x48).
// ---------------------------------------------------------------------------
__global__ void __launch_bounds__(256)
agg_transform_kernel(const float* __restrict__ s_feats, const int* __restrict__ nbr,
                     const float* __restrict__ Wv, const float* __restrict__ gate,
                     float* __restrict__ out) {
    __shared__ __align__(32) _Float16 sWv[24 * 32 * FS];  // A: [192][64], 12 rt x 2 s
    __shared__ __align__(32) _Float16 sAgg[6 * 32 * FS];  // B: [64][48],  3 ct x 2 s

    const int tid  = threadIdx.x;
    const int lane = tid & 31;
    const int w    = tid >> 5;
    const int n0   = blockIdx.x * 16;

    for (int i = tid; i < 192 * 64; i += 256) {
        int o = i >> 6, dd = i & 63;
        sWv[frag_addr(o, dd, 2)] = (_Float16)Wv[i];
    }

    // Gather + mean over 32 neighbors: 3072 accumulators, 12 per thread
#pragma unroll
    for (int j = 0; j < 12; ++j) {
        int flat = tid + 256 * j;
        int nl = flat / 192, t = flat - nl * 192;
        const int* np = nbr + (size_t)(n0 + nl) * KNBR;
        float acc = 0.f;
#pragma unroll 4
        for (int k = 0; k < KNBR; ++k) {
            if (k + 8 < KNBR)
                __builtin_prefetch(&s_feats[(size_t)np[k + 8] * 192 + t], 0, 0);
            acc += s_feats[(size_t)np[k] * 192 + t];
        }
        int dd = t / 3, tt = t - dd * 3;
        int col = nl * 3 + tt;                       // B element (k=dd, n=col)
        sAgg[frag_addr(col, dd, 2)] = (_Float16)(acc * (1.f / 32.f));
    }
    __syncthreads();

    // 12 row tiles x 3 col tiles, K=64 (2 slices)
    for (int tile = w; tile < 36; tile += 8) {
        int rt = tile % 12, ct = tile / 12;
        v8f acc = {};
#pragma unroll
        for (int s = 0; s < 2; ++s)
            acc = wmma_f16(load_frag(sWv, rt, s, 2, lane),
                           load_frag(sAgg, ct, s, 2, lane), acc);
        int col = ct * 16 + (lane & 15);
        int rb  = rt * 16 + ((lane < 16) ? 0 : 8);
        int nl  = col / 3, tt = col - nl * 3;
        int n   = n0 + nl;
#pragma unroll
        for (int i = 0; i < 8; ++i) {
            int o = rb + i;
            out[(size_t)n * (OUT * 3) + o * 3 + tt] = acc[i] * gate[(size_t)n * OUT + o];
        }
    }
}

extern "C" void kernel_launch(void* const* d_in, const int* in_sizes, int n_in,
                              void* d_out, int out_size, void* d_ws, size_t ws_size,
                              hipStream_t stream) {
    const float* q_pts   = (const float*)d_in[0];
    const float* s_pts   = (const float*)d_in[1];
    const float* s_feats = (const float*)d_in[2];
    const int*   nbr     = (const int*)d_in[3];
    const float* normals = (const float*)d_in[4];
    const float* W1      = (const float*)d_in[5];
    const float* b1      = (const float*)d_in[6];
    const float* W2      = (const float*)d_in[7];
    const float* b2      = (const float*)d_in[8];
    const float* W3      = (const float*)d_in[9];
    const float* b3      = (const float*)d_in[10];
    const float* Wg      = (const float*)d_in[11];
    const float* bg      = (const float*)d_in[12];
    const float* Wv      = (const float*)d_in[13];
    float* out     = (float*)d_out;
    float* gate_ws = (float*)d_ws;   // N*192 f32 = 15.36 MB scratch

    ppf_mlp_gate_kernel<<<(NPTS + 7) / 8, 256, 0, stream>>>(
        q_pts, s_pts, nbr, normals, W1, b1, W2, b2, W3, b3, Wg, bg, gate_ws);
    agg_transform_kernel<<<(NPTS + 15) / 16, 256, 0, stream>>>(
        s_feats, nbr, Wv, gate_ws, out);
}
